// EdgeClassifier_5798205850124
// MI455X (gfx1250) — matrix operand order repaired and checked by reading, not compiled, not run on hardware
//
#include <hip/hip_runtime.h>
#include <hip/hip_bf16.h>

typedef __attribute__((ext_vector_type(16))) __bf16 vbf16x16;
typedef __attribute__((ext_vector_type(8)))  float   v8f;

#define IN_FEATS 128
#define HIDDEN   32
#define W1_COLS  (2 * IN_FEATS)
// fragment table: 16 fragments (2 n-tiles x 8 k-chunks) x 32 lanes x 16 bf16 = 16 KB
#define FRAG_ELEMS   (32 * 16)
#define WTAB_ELEMS   (16 * FRAG_ELEMS)
#define WTAB_BYTES   (WTAB_ELEMS * 2)
#define EMB_WS_OFF   WTAB_BYTES              // bf16 embedding table follows the frag table
#define HROW_STRIDE  33                      // padded 16x32 h tile row stride
#define HTILE_ELEMS  (16 * HROW_STRIDE)

// Pack 8 consecutive f32 (two float4) into elements [base, base+8) of a bf16x16 frag.
__device__ __forceinline__ void pack8(vbf16x16& v, int base, const float4& x, const float4& y) {
    v[base + 0] = (__bf16)x.x; v[base + 1] = (__bf16)x.y;
    v[base + 2] = (__bf16)x.z; v[base + 3] = (__bf16)x.w;
    v[base + 4] = (__bf16)y.x; v[base + 5] = (__bf16)y.y;
    v[base + 6] = (__bf16)y.z; v[base + 7] = (__bf16)y.w;
}

// Build one A fragment (16x32 bf16 tile, this lane's share) from a feature row.
// A layout: lane L holds row m=L&15; elems 0..7 = K[kh*8..+7], elems 8..15 = K[16+kh*8..+7].
template <bool BF16E>
__device__ __forceinline__ vbf16x16 load_a(const void* rowv, int col, int kh) {
    if constexpr (BF16E) {
        const __bf16* row = (const __bf16*)rowv;      // bf16 row: direct 2x b128, no cvt
        union { vbf16x16 v; uint4 q[2]; } u;
        u.q[0] = *(const uint4*)(row + col + kh * 8);
        u.q[1] = *(const uint4*)(row + col + 16 + kh * 8);
        return u.v;
    } else {
        const float* row = (const float*)rowv;        // f32 row: 4x b128 + cvt
        const float4* p0 = (const float4*)(row + col + kh * 8);
        const float4* p1 = (const float4*)(row + col + 16 + kh * 8);
        float4 a0 = p0[0], a1 = p0[1];
        float4 a2 = p1[0], a3 = p1[1];
        vbf16x16 A;
        pack8(A, 0, a0, a1);
        pack8(A, 8, a2, a3);
        return A;
    }
}

// ---- One-wave prep: convert W1 (f32 [32,256]) into per-lane WMMA B fragments (bf16). ----
// B layout (32x16 bf16): lane L holds column n = L&15; elems 0..7 = K[kh*8..+7], elems 8..15 = K[16+kh*8..+7].
__global__ __launch_bounds__(32) void pack_w1_frags(const float* __restrict__ W1,
                                                    __bf16* __restrict__ wfrag) {
    const int lane = threadIdx.x & 31;
    const int m    = lane & 15;
    const int kh   = lane >> 4;
#pragma unroll
    for (int nt = 0; nt < 2; ++nt) {
#pragma unroll
        for (int ch = 0; ch < 8; ++ch) {
            const float* wrow = W1 + (size_t)(nt * 16 + m) * W1_COLS + ch * 32 + kh * 8;
            const float4* wp  = (const float4*)wrow;
            float4 w0 = wp[0], w1 = wp[1], w2 = wp[4], w3 = wp[5];
            vbf16x16 f;
            pack8(f, 0, w0, w1);
            pack8(f, 8, w2, w3);
            const int frag = nt * 8 + ch;
            *(vbf16x16*)(wfrag + (size_t)frag * FRAG_ELEMS + lane * 16) = f;
        }
    }
}

// ---- Streaming prep: node_embed f32 -> bf16 (8 elements per thread-iteration). ----
__global__ __launch_bounds__(256) void cvt_embed_bf16(const float* __restrict__ in,
                                                      __bf16* __restrict__ outb,
                                                      long n8) {
    long i = (long)blockIdx.x * blockDim.x + threadIdx.x;
    const long stride = (long)gridDim.x * blockDim.x;
    for (; i < n8; i += stride) {
        const float4* p = (const float4*)(in + i * 8);
        float4 a = p[0], b = p[1];
        union { uint4 q; __bf16 h[8]; } u;
        u.h[0] = (__bf16)a.x; u.h[1] = (__bf16)a.y;
        u.h[2] = (__bf16)a.z; u.h[3] = (__bf16)a.w;
        u.h[4] = (__bf16)b.x; u.h[5] = (__bf16)b.y;
        u.h[6] = (__bf16)b.z; u.h[7] = (__bf16)b.w;
        *(uint4*)(outb + i * 8) = u.q;
    }
}

template <bool BF16E>
__global__ __launch_bounds__(256) void edge_mlp_wmma(
    const int*    __restrict__ src,
    const int*    __restrict__ dst,
    const void*   __restrict__ embed,   // [N_NODES, 128] f32 or bf16
    const __bf16* __restrict__ wfrag,   // prepacked B fragments (16 KB)
    const float*  __restrict__ b1,      // [32]
    const float*  __restrict__ W2,      // [2, 32]
    const float*  __restrict__ b2,      // [2]
    float*        __restrict__ out,     // [N_EDGES, 2]
    int n_edges)
{
    __shared__ uint4 wtab4[WTAB_BYTES / 16];            // 16 KB bf16 fragment table
    __shared__ float h_lds[8 * 2 * HTILE_ELEMS];        // 2 h tiles per wave

    // ---- Stage weight fragments into LDS once per block ----
    {
        const uint4* g4 = (const uint4*)wfrag;
#pragma unroll
        for (int i = 0; i < 4; ++i)
            wtab4[threadIdx.x + i * 256] = g4[threadIdx.x + i * 256];
    }
    __syncthreads();
    const __bf16* wtab = (const __bf16*)wtab4;

    const int lane = threadIdx.x & 31;
    const int wave = threadIdx.x >> 5;
    const int m    = lane & 15;   // A row / B column / C column within 16-wide tile
    const int kh   = lane >> 4;   // which K-half this lane holds

    // Each wave handles 32 edges = 2 A tiles -> 4 independent WMMA accumulator chains.
    const long base = ((long)blockIdx.x * 8 + wave) * 32;

    // ---- Gather edge endpoints for both tiles ----
    long e0 = base + m;       if (e0 >= n_edges) e0 = n_edges - 1;
    long e1 = base + 16 + m;  if (e1 >= n_edges) e1 = n_edges - 1;
    const size_t rowbytes = BF16E ? (IN_FEATS * 2) : (IN_FEATS * 4);
    const char* eb = (const char*)embed;
    const void* s0 = eb + (size_t)src[e0] * rowbytes;
    const void* d0 = eb + (size_t)dst[e0] * rowbytes;
    const void* s1 = eb + (size_t)src[e1] * rowbytes;
    const void* d1 = eb + (size_t)dst[e1] * rowbytes;

    // ---- Layer 1: two 16x32 C tiles over K=256, 32 bf16 WMMAs per wave ----
    v8f c00 = {}, c01 = {};   // tile 0, hidden n 0..15 / 16..31
    v8f c10 = {}, c11 = {};   // tile 1
#pragma unroll
    for (int ch = 0; ch < 8; ++ch) {
        const void* r0 = (ch < 4) ? s0 : d0;
        const void* r1 = (ch < 4) ? s1 : d1;
        const int col = (ch & 3) * 32;
        vbf16x16 A0 = load_a<BF16E>(r0, col, kh);
        vbf16x16 A1 = load_a<BF16E>(r1, col, kh);
        // B fragments from LDS; each feeds two WMMAs
        vbf16x16 B0 = *(const vbf16x16*)(wtab + (0 * 8 + ch) * FRAG_ELEMS + lane * 16);
        vbf16x16 B1 = *(const vbf16x16*)(wtab + (1 * 8 + ch) * FRAG_ELEMS + lane * 16);
        c00 = __builtin_amdgcn_wmma_f32_16x16x32_bf16(false, A0, false, B0, (short)0, c00, false, false);
        c10 = __builtin_amdgcn_wmma_f32_16x16x32_bf16(false, A1, false, B0, (short)0, c10, false, false);
        c01 = __builtin_amdgcn_wmma_f32_16x16x32_bf16(false, A0, false, B1, (short)0, c01, false, false);
        c11 = __builtin_amdgcn_wmma_f32_16x16x32_bf16(false, A1, false, B1, (short)0, c11, false, false);
    }

    // ---- bias + ReLU, scatter C fragments into the two h tiles in LDS ----
    const float bias0 = b1[m];
    const float bias1 = b1[16 + m];
    float* h0 = &h_lds[(wave * 2 + 0) * HTILE_ELEMS];
    float* h1 = &h_lds[(wave * 2 + 1) * HTILE_ELEMS];
#pragma unroll
    for (int r = 0; r < 8; ++r) {
        const int M = r + 8 * kh;                 // edge row 0..15 within tile
        float v00 = c00[r] + bias0; v00 = v00 > 0.0f ? v00 : 0.0f;
        float v01 = c01[r] + bias1; v01 = v01 > 0.0f ? v01 : 0.0f;
        float v10 = c10[r] + bias0; v10 = v10 > 0.0f ? v10 : 0.0f;
        float v11 = c11[r] + bias1; v11 = v11 > 0.0f ? v11 : 0.0f;
        h0[M * HROW_STRIDE + m]      = v00;
        h0[M * HROW_STRIDE + 16 + m] = v01;
        h1[M * HROW_STRIDE + m]      = v10;
        h1[M * HROW_STRIDE + 16 + m] = v11;
    }
    __syncthreads();

    // ---- Layer 2: 32 lanes -> (16 edges x 2 classes) per tile, 2 tiles ----
    const int em  = lane >> 1;     // edge within tile
    const int cls = lane & 1;      // output class
    const float* w2 = W2 + cls * HIDDEN;
    const float  bb = b2[cls];
#pragma unroll
    for (int t = 0; t < 2; ++t) {
        const long eidx = base + t * 16 + em;
        if (eidx < n_edges) {
            const float* ht = (t == 0) ? h0 : h1;
            float acc = bb;
#pragma unroll
            for (int n = 0; n < HIDDEN; ++n)
                acc += ht[em * HROW_STRIDE + n] * w2[n];
            out[eidx * 2 + cls] = acc;
        }
    }
}

extern "C" void kernel_launch(void* const* d_in, const int* in_sizes, int n_in,
                              void* d_out, int out_size, void* d_ws, size_t ws_size,
                              hipStream_t stream) {
    (void)n_in; (void)out_size;
    const int*   src   = (const int*)  d_in[0];
    const int*   dst   = (const int*)  d_in[1];
    const float* embed = (const float*)d_in[2];
    const float* W1    = (const float*)d_in[3];
    const float* b1    = (const float*)d_in[4];
    const float* W2    = (const float*)d_in[5];
    const float* b2    = (const float*)d_in[6];
    float*       out   = (float*)d_out;

    __bf16* wfrag   = (__bf16*)d_ws;                              // 16 KB
    __bf16* emb_bf  = (__bf16*)((char*)d_ws + EMB_WS_OFF);        // N_NODES*128 bf16

    const int  n_edges     = in_sizes[0];
    const long embed_elems = in_sizes[2];
    const int  n_pairs = (n_edges + 31) / 32;      // 32 edges per wave
    const int  blocks  = (n_pairs + 7) / 8;        // 8 waves per 256-thread block

    pack_w1_frags<<<1, 32, 0, stream>>>(W1, wfrag);

    const size_t need = (size_t)EMB_WS_OFF + (size_t)embed_elems * 2;
    if (ws_size >= need) {
        // Pre-convert embeddings to bf16: halves gather bytes and removes all
        // A-side cvts from the WMMA loop.
        const long n8 = embed_elems / 8;
        int cblocks = (int)((n8 + 255) / 256);
        if (cblocks > 4096) cblocks = 4096;
        cvt_embed_bf16<<<cblocks, 256, 0, stream>>>(embed, emb_bf, n8);
        edge_mlp_wmma<true><<<blocks, 256, 0, stream>>>(src, dst, (const void*)emb_bf,
                                                        wfrag, b1, W2, b2, out, n_edges);
    } else {
        edge_mlp_wmma<false><<<blocks, 256, 0, stream>>>(src, dst, (const void*)embed,
                                                         wfrag, b1, W2, b2, out, n_edges);
    }
}
